// TransitionModel_78563541778492
// MI455X (gfx1250) — matrix operand order repaired and checked by reading, not compiled, not run on hardware
//
#include <hip/hip_runtime.h>
#include <hip/hip_bf16.h>
#include <math.h>

#define T_STEPS 49
#define B_ROWS  2048
#define S_DIM   30
#define A_DIM   6
#define BE_DIM  200
#define E_DIM   1024
#define MIN_STD 0.1f

typedef __attribute__((ext_vector_type(16))) __bf16 v16bf;
typedef __attribute__((ext_vector_type(8)))  float  v8f;

union ABuf { v16bf v; __bf16 u[16]; };

__device__ __forceinline__ v8f wmma_bf16(v16bf a, v16bf b, v8f c) {
  return __builtin_amdgcn_wmma_f32_16x16x32_bf16(false, a, false, b, (short)0, c,
                                                 false, false);
}

// ---- packed fragment layout: P[(tile*kcTotal + kc)*32 + lane][16 halves] ----
// A-fragment element (lane,i): M=lane&15, K = kc*32 + ((i>>3)<<4) + (lane>>4)*8 + (i&7)
// B-fragment element (lane,i): N=lane&15, K = kc*32 + ((i>>3)<<4) + (lane>>4)*8 + (i&7)
__device__ __forceinline__ v16bf load_pk(const __bf16* __restrict__ P,
                                         int tile, int kcTotal, int kc, int lane) {
  return *(const v16bf*)(P + ((size_t)(tile * kcTotal + kc) * 32 + lane) * 16);
}

// scatter one f32 value into a packed A-fragment buffer (native bf16 cvt)
__device__ __forceinline__ void store_apk(__bf16* __restrict__ P,
                                          int kcTotal, int row, int col, float v) {
  int rt = row >> 4, kr = col & 31, kc = col >> 5;
  int half = (kr >> 3) & 1;
  int i = ((kr >> 4) << 3) | (kr & 7);
  int lane = (row & 15) + (half << 4);
  P[((size_t)(rt * kcTotal + kc) * 32 + lane) * 16 + i] = (__bf16)v;
}

__device__ __forceinline__ void store_lds_pk(__bf16* P, int lrow, int col, float v) {
  int kr = col & 31, kc = col >> 5;
  int half = (kr >> 3) & 1;
  int i = ((kr >> 4) << 3) | (kr & 7);
  int lane = lrow + (half << 4);
  P[((size_t)kc * 32 + lane) * 16 + i] = (__bf16)v;
}

// ---------------- prep: zero pad regions ----------------
__global__ void k_zero(unsigned int* __restrict__ p, unsigned int n) {
  unsigned int i = blockIdx.x * blockDim.x + threadIdx.x;
  if (i < n) p[i] = 0u;
}

// ---------------- prep: pack f32 weights [K,*] column-slice -> B fragments ---
__global__ void k_pack_w(const float* __restrict__ W, int ldw, int Ksrc, int Ncols,
                         int kcTotal, int kcBase, __bf16* __restrict__ dst) {
  int lane = threadIdx.x;                     // 32 threads
  int ct = blockIdx.x, kc = blockIdx.y;
  int half = lane >> 4;
  int col = ct * 16 + (lane & 15);
  int cc = min(col, Ncols - 1);
  ABuf b;
#pragma unroll
  for (int i = 0; i < 16; ++i) {
    int k = kc * 32 + ((i >> 3) << 4) + (half << 3) + (i & 7);
    int kk = min(k, Ksrc - 1);
    float v = W[(size_t)kk * ldw + cc];       // always-valid load, no branch
    b.u[i] = (__bf16)((k < Ksrc && col < Ncols) ? v : 0.f);
  }
  *(v16bf*)(dst + ((size_t)(ct * kcTotal + kcBase + kc) * 32 + lane) * 16) = b.v;
}

// ------- prep/per-step: pack f32 activations [B,Ksrc] -> A fragments ---------
// Optionally prefetches the next step's source slab (global_prefetch_b8).
__global__ void k_pack_a(const float* __restrict__ A,
                         const float* __restrict__ Apf,   // next-step slab or null
                         int Ksrc, int kcTotal, __bf16* __restrict__ dst) {
  int lane = threadIdx.x;                     // 32 threads
  int rt = blockIdx.x, kc = blockIdx.y;
  int half = lane >> 4;
  int row = rt * 16 + (lane & 15);
  ABuf a;
#pragma unroll
  for (int i = 0; i < 16; ++i) {
    int k = kc * 32 + ((i >> 3) << 4) + (half << 3) + (i & 7);
    int kk = min(k, Ksrc - 1);
    float v = A[(size_t)row * Ksrc + kk];
    a.u[i] = (__bf16)((k < Ksrc) ? v : 0.f);
  }
  *(v16bf*)(dst + ((size_t)(rt * kcTotal + kc) * 32 + lane) * 16) = a.v;
  if (Apf)   // warm L2 for the next timestep's HBM-resident stream
    __builtin_prefetch(Apf + (size_t)row * Ksrc + (size_t)kc * 32, 0, 1);
}

// ---------------- sa = relu([post*nt | act] @ W_sa + b_sa) -> packed sa_pk ---
__global__ void k_sa(const float* __restrict__ post_prev, // [B,30]
                     const float* __restrict__ nt,        // [B] or null
                     const float* __restrict__ act,       // [B,6]
                     const __bf16* __restrict__ wsa_pk,   // 13x2 chunks
                     const float* __restrict__ b_sa,      // [200]
                     __bf16* __restrict__ sa_pk) {
  int lane = threadIdx.x & 31, wave = threadIdx.x >> 5;
  int ct = blockIdx.y * 4 + wave;
  if (ct >= 13) return;
  int row0 = blockIdx.x * 16;
  int half = lane >> 4;
  int mrow = row0 + (lane & 15);
  int col  = ct * 16 + (lane & 15);
  bool ok  = col < BE_DIM;
  float ntv = nt ? nt[mrow] : 1.0f;
  v8f acc = {};
#pragma unroll
  for (int kc = 0; kc < 2; ++kc) {
    ABuf a;
#pragma unroll
    for (int i = 0; i < 16; ++i) {
      int k = kc * 32 + ((i >> 3) << 4) + (half << 3) + (i & 7);
      float vp = post_prev[mrow * S_DIM + min(k, S_DIM - 1)] * ntv;
      float va = act[mrow * A_DIM + min(max(k - S_DIM, 0), A_DIM - 1)];
      a.u[i] = (__bf16)((k < S_DIM) ? vp : ((k < S_DIM + A_DIM) ? va : 0.f));
    }
    v16bf b = load_pk(wsa_pk, ct, 2, kc, lane);
    acc = wmma_bf16(a.v, b, acc);
  }
  if (ok) {
    float bias = b_sa[col];
#pragma unroll
    for (int r = 0; r < 8; ++r)
      store_apk(sa_pk, 7, row0 + half * 8 + r, col, fmaxf(acc[r] + bias, 0.f));
  }
}

// ---------------- GRU: 6 gate GEMMs fused; pure fragment loads ----------------
__global__ void k_gru(const __bf16* __restrict__ sa_pk,
                      const __bf16* __restrict__ belp_pk,
                      const float* __restrict__ bel_prev_f32,  // [B,200]
                      const __bf16* __restrict__ wir,
                      const __bf16* __restrict__ wiz,
                      const __bf16* __restrict__ win,
                      const __bf16* __restrict__ whr,
                      const __bf16* __restrict__ whz,
                      const __bf16* __restrict__ whn,
                      const float* __restrict__ bi,        // [600]
                      const float* __restrict__ bhn,       // [200]
                      float* __restrict__ bel_out_f32,     // [B,200]
                      __bf16* __restrict__ bel_out_pk) {
  int lane = threadIdx.x & 31, wave = threadIdx.x >> 5;
  int ct = blockIdx.y * 4 + wave;
  if (ct >= 13) return;
  int rt = blockIdx.x, row0 = rt * 16;
  int half = lane >> 4;
  int col  = ct * 16 + (lane & 15);
  bool ok  = col < BE_DIM;
  v8f air = {}, aiz = {}, ain = {}, ahr = {}, ahz = {}, ahn = {};
  for (int kc = 0; kc < 7; ++kc) {
    v16bf a1 = load_pk(sa_pk,   rt, 7, kc, lane);
    v16bf a2 = load_pk(belp_pk, rt, 7, kc, lane);
    air = wmma_bf16(a1, load_pk(wir, ct, 7, kc, lane), air);
    aiz = wmma_bf16(a1, load_pk(wiz, ct, 7, kc, lane), aiz);
    ain = wmma_bf16(a1, load_pk(win, ct, 7, kc, lane), ain);
    ahr = wmma_bf16(a2, load_pk(whr, ct, 7, kc, lane), ahr);
    ahz = wmma_bf16(a2, load_pk(whz, ct, 7, kc, lane), ahz);
    ahn = wmma_bf16(a2, load_pk(whn, ct, 7, kc, lane), ahn);
  }
  if (ok) {
    float br_ = bi[col], bz_ = bi[col + 200], bn_ = bi[col + 400], bh = bhn[col];
#pragma unroll
    for (int r = 0; r < 8; ++r) {
      int row = row0 + half * 8 + r;
      float rg = 1.f / (1.f + __expf(-(air[r] + br_ + ahr[r])));
      float zg = 1.f / (1.f + __expf(-(aiz[r] + bz_ + ahz[r])));
      float ng = tanhf(ain[r] + bn_ + rg * (ahn[r] + bh));
      float nb = (1.f - zg) * ng + zg * bel_prev_f32[row * BE_DIM + col];
      bel_out_f32[row * BE_DIM + col] = nb;
      store_apk(bel_out_pk, 7, row, col, nb);
    }
  }
}

// ---- fused 2-layer head; stage1 -> LDS bf16 fragments -> stage2 --------------
__global__ void __launch_bounds__(416)
k_head(const __bf16* __restrict__ belpk,
       const __bf16* __restrict__ obspk,   // null for prior head
       const __bf16* __restrict__ W1pk, int kcT1,
       const float* __restrict__ b1,               // [200]
       const __bf16* __restrict__ W2m,     // 2 tiles x 7 chunks
       const __bf16* __restrict__ W2s,
       const float* __restrict__ b2,               // [60]
       const float* __restrict__ eps,              // [B,30]
       float* __restrict__ o_state, float* __restrict__ o_mean,
       float* __restrict__ o_std) {
  __shared__ __align__(32) __bf16 hid_pk[7 * 512];   // 16 x 224 bf16 fragments
  int lane = threadIdx.x & 31, wave = threadIdx.x >> 5;     // 13 waves
  int rt = blockIdx.x, row0 = rt * 16;
  int half = lane >> 4;

  // stage 1: hid = relu([bel|obs] @ W1 + b1), one wave per 16-col tile
  {
    int col = wave * 16 + (lane & 15);
    bool ok = col < BE_DIM;
    v8f acc = {};
    for (int kc = 0; kc < 7; ++kc) {
      v16bf a = load_pk(belpk, rt, 7, kc, lane);
      v16bf b = load_pk(W1pk, wave, kcT1, kc, lane);
      acc = wmma_bf16(a, b, acc);
    }
    if (obspk) {
      for (int kc = 0; kc < 32; ++kc) {
        v16bf a = load_pk(obspk, rt, 32, kc, lane);
        v16bf b = load_pk(W1pk, wave, kcT1, 7 + kc, lane);
        acc = wmma_bf16(a, b, acc);
      }
    }
    float bias = ok ? b1[col] : 0.f;
#pragma unroll
    for (int r = 0; r < 8; ++r) {
      float v = ok ? fmaxf(acc[r] + bias, 0.f) : 0.f;
      store_lds_pk(hid_pk, half * 8 + r, col, v);
    }
    if (wave == 12) {                 // zero-fill K pad cols 208..223
#pragma unroll
      for (int r = 0; r < 8; ++r) store_lds_pk(hid_pk, half * 8 + r, col + 16, 0.f);
    }
  }
  __syncthreads();

  // stage 2: waves 0..1 compute paired mean/std tiles from LDS fragments
  if (wave < 2) {
    int cm = wave * 16 + (lane & 15);      // mean col; std col = cm + 30
    bool ok = cm < S_DIM;
    v8f am = {}, as_ = {};
    for (int kc = 0; kc < 7; ++kc) {
      v16bf a = *(const v16bf*)&hid_pk[((size_t)kc * 32 + lane) * 16];
      am  = wmma_bf16(a, load_pk(W2m, wave, 7, kc, lane), am);
      as_ = wmma_bf16(a, load_pk(W2s, wave, 7, kc, lane), as_);
    }
    if (ok) {
      float bm_ = b2[cm], bs_ = b2[cm + S_DIM];
#pragma unroll
      for (int r = 0; r < 8; ++r) {
        int row = row0 + half * 8 + r;
        float mean = am[r] + bm_;
        float x = as_[r] + bs_;
        float sp = fmaxf(x, 0.f) + log1pf(__expf(-fabsf(x))) + MIN_STD;
        o_mean [row * S_DIM + cm] = mean;
        o_std  [row * S_DIM + cm] = sp;
        o_state[row * S_DIM + cm] = mean + sp * eps[row * S_DIM + cm];
      }
    }
  }
}

extern "C" void kernel_launch(void* const* d_in, const int* in_sizes, int n_in,
                              void* d_out, int out_size, void* d_ws, size_t ws_size,
                              hipStream_t stream) {
  (void)in_sizes; (void)n_in; (void)out_size; (void)ws_size;
  const float* prev_state   = (const float*)d_in[0];
  const float* actions      = (const float*)d_in[1];
  const float* prev_belief  = (const float*)d_in[2];
  const float* observations = (const float*)d_in[3];
  const float* nonterminals = (const float*)d_in[4];
  const float* noise_prior  = (const float*)d_in[5];
  const float* noise_post   = (const float*)d_in[6];
  const float* W_sa = (const float*)d_in[7];
  const float* b_sa = (const float*)d_in[8];
  const float* Wi   = (const float*)d_in[9];
  const float* bi   = (const float*)d_in[10];
  const float* Wh   = (const float*)d_in[11];
  const float* bhn  = (const float*)d_in[12];
  const float* W_bp = (const float*)d_in[13];
  const float* b_bp = (const float*)d_in[14];
  const float* W_sp = (const float*)d_in[15];
  const float* b_sp = (const float*)d_in[16];
  const float* W_bq = (const float*)d_in[17];
  const float* b_bq = (const float*)d_in[18];
  const float* W_sq = (const float*)d_in[19];
  const float* b_sq = (const float*)d_in[20];

  float* out = (float*)d_out;
  const size_t TB = (size_t)T_STEPS * B_ROWS;
  float* o_bel   = out;                      // [T,B,200]
  float* o_prior = o_bel   + TB * BE_DIM;    // [T,B,30] x6
  float* o_pm    = o_prior + TB * S_DIM;
  float* o_ps    = o_pm    + TB * S_DIM;
  float* o_post  = o_ps    + TB * S_DIM;
  float* o_qm    = o_post  + TB * S_DIM;
  float* o_qs    = o_qm    + TB * S_DIM;

  // ---- workspace carving (units: __bf16) ----
  __bf16* ws = (__bf16*)d_ws;
  const size_t SA_PK = 128 * 7 * 512;        // 458752
  const size_t OB_PK = 128 * 32 * 512;       // 2097152
  const size_t WG    = 13 * 7 * 512;         // 46592
  size_t off = 0;
  __bf16* sa_pk = ws + off; off += SA_PK;
  __bf16* bpk0  = ws + off; off += SA_PK;
  __bf16* bpk1  = ws + off; off += SA_PK;    // first 3 zeroed below
  __bf16* binit = ws + off; off += SA_PK;
  __bf16* obspk = ws + off; off += OB_PK;
  __bf16* wsa   = ws + off; off += 13 * 2 * 512;
  __bf16* wir   = ws + off; off += WG;
  __bf16* wiz   = ws + off; off += WG;
  __bf16* win   = ws + off; off += WG;
  __bf16* whr   = ws + off; off += WG;
  __bf16* whz   = ws + off; off += WG;
  __bf16* whn   = ws + off; off += WG;
  __bf16* wbp   = ws + off; off += WG;
  __bf16* wbq   = ws + off; off += 13 * 39 * 512;
  __bf16* wspm  = ws + off; off += 2 * 7 * 512;
  __bf16* wsps  = ws + off; off += 2 * 7 * 512;
  __bf16* wsqm  = ws + off; off += 2 * 7 * 512;
  __bf16* wsqs  = ws + off; off += 2 * 7 * 512;

  // ---- one-time prep: zero activation pads, pack weights + initial belief ----
  {
    unsigned int nz = (unsigned int)(3 * SA_PK / 2);      // as u32 words
    k_zero<<<dim3((nz + 255) / 256), dim3(256), 0, stream>>>((unsigned int*)ws, nz);
    dim3 b32(32);
    k_pack_w<<<dim3(13, 2),  b32, 0, stream>>>(W_sa,        200,  36, 200,  2, 0, wsa);
    k_pack_w<<<dim3(13, 7),  b32, 0, stream>>>(Wi,          600, 200, 200,  7, 0, wir);
    k_pack_w<<<dim3(13, 7),  b32, 0, stream>>>(Wi + 200,    600, 200, 200,  7, 0, wiz);
    k_pack_w<<<dim3(13, 7),  b32, 0, stream>>>(Wi + 400,    600, 200, 200,  7, 0, win);
    k_pack_w<<<dim3(13, 7),  b32, 0, stream>>>(Wh,          600, 200, 200,  7, 0, whr);
    k_pack_w<<<dim3(13, 7),  b32, 0, stream>>>(Wh + 200,    600, 200, 200,  7, 0, whz);
    k_pack_w<<<dim3(13, 7),  b32, 0, stream>>>(Wh + 400,    600, 200, 200,  7, 0, whn);
    k_pack_w<<<dim3(13, 7),  b32, 0, stream>>>(W_bp,        200, 200, 200,  7, 0, wbp);
    k_pack_w<<<dim3(13, 7),  b32, 0, stream>>>(W_bq,        200, 200, 200, 39, 0, wbq);
    k_pack_w<<<dim3(13, 32), b32, 0, stream>>>(W_bq + 200 * 200,
                                                            200, 1024, 200, 39, 7, wbq);
    k_pack_w<<<dim3(2, 7),   b32, 0, stream>>>(W_sp,         60, 200,  30,  7, 0, wspm);
    k_pack_w<<<dim3(2, 7),   b32, 0, stream>>>(W_sp + 30,    60, 200,  30,  7, 0, wsps);
    k_pack_w<<<dim3(2, 7),   b32, 0, stream>>>(W_sq,         60, 200,  30,  7, 0, wsqm);
    k_pack_w<<<dim3(2, 7),   b32, 0, stream>>>(W_sq + 30,    60, 200,  30,  7, 0, wsqs);
    k_pack_a<<<dim3(128, 7), b32, 0, stream>>>(prev_belief, observations,
                                               200, 7, binit);
  }

  dim3 gGemm(B_ROWS / 16, 4), bGemm(128);
  dim3 gHead(B_ROWS / 16),    bHead(13 * 32);
  __bf16* bpk[2] = { bpk0, bpk1 };

  for (int t = 0; t < T_STEPS; ++t) {
    const float* post_prev = (t == 0) ? prev_state
                                      : (o_post + (size_t)(t - 1) * B_ROWS * S_DIM);
    const float* nt        = (t == 0) ? nullptr
                                      : (nonterminals + (size_t)(t - 1) * B_ROWS);
    const float* act_t     = actions + (size_t)t * B_ROWS * A_DIM;
    const float* belp_f32  = (t == 0) ? prev_belief
                                      : (o_bel + (size_t)(t - 1) * B_ROWS * BE_DIM);
    const __bf16* belp_pk = (t == 0) ? binit : bpk[(t - 1) & 1];
    __bf16* belo_pk = bpk[t & 1];
    float* bel_t = o_bel + (size_t)t * B_ROWS * BE_DIM;
    const float* obs_t = observations + (size_t)t * B_ROWS * E_DIM;
    const float* obs_n = (t + 1 < T_STEPS)
                       ? observations + (size_t)(t + 1) * B_ROWS * E_DIM : nullptr;
    const size_t so = (size_t)t * B_ROWS * S_DIM;

    k_pack_a<<<dim3(128, 32), dim3(32), 0, stream>>>(obs_t, obs_n, E_DIM, 32, obspk);
    k_sa<<<gGemm, bGemm, 0, stream>>>(post_prev, nt, act_t, wsa, b_sa, sa_pk);
    k_gru<<<gGemm, bGemm, 0, stream>>>(sa_pk, belp_pk, belp_f32,
                                       wir, wiz, win, whr, whz, whn,
                                       bi, bhn, bel_t, belo_pk);
    k_head<<<gHead, bHead, 0, stream>>>(belo_pk, nullptr, wbp, 7, b_bp,
                                        wspm, wsps, b_sp, noise_prior + so,
                                        o_prior + so, o_pm + so, o_ps + so);
    k_head<<<gHead, bHead, 0, stream>>>(belo_pk, obspk, wbq, 39, b_bq,
                                        wsqm, wsqs, b_sq, noise_post + so,
                                        o_post + so, o_qm + so, o_qs + so);
  }
}
